// ESN_26439818674921
// MI455X (gfx1250) — compile-verified, matrix operand order checked by reading, not run on hardware
//
#include <hip/hip_runtime.h>

// ESN recurrence on MI455X (gfx1250, wave32, WMMA):
//   s_t = tanh(x_t @ Win^T + s_{t-1} @ W^T),  output = s_{T-1}  [B=256, R=2048] f32
//
//  - Win, W, X converted to f16 in d_ws; X transposed to [T, B, IN].
//  - 512 step kernels, ping-ponging two f16 state buffers.
//  - Step kernel: f16 inputs, f32 accumulation via v_wmma_f32_16x16x32_f16.
//    Block = 256 threads = 8 waves (4 M-waves x 2 N-waves); wave tile 16x64;
//    block tile 64x128; grid = (16, 4). Stream pointers hoisted so the K-loop
//    uses immediate-offset global_load_b128 only.

#define BB   256
#define TT   512
#define NIN  64
#define RR   2048

typedef __attribute__((ext_vector_type(16))) _Float16 v16h;
typedef __attribute__((ext_vector_type(8)))  _Float16 v8h;
typedef __attribute__((ext_vector_type(8)))  float    v8f;

// ---------------- prep kernels ----------------

__global__ void esn_cvt_f16(const float* __restrict__ in, _Float16* __restrict__ out, int n) {
    int i = blockIdx.x * blockDim.x + threadIdx.x;
    if (i < n) out[i] = (_Float16)in[i];
}

// X [B, T, IN] f32  ->  Xb [T, B, IN] f16
__global__ void esn_transpose_x(const float* __restrict__ X, _Float16* __restrict__ Xb, int n) {
    int i = blockIdx.x * blockDim.x + threadIdx.x;   // flat index into Xb
    if (i >= n) return;
    int in_i = i % NIN;
    int rest = i / NIN;
    int b    = rest % BB;
    int t    = rest / BB;
    Xb[i] = (_Float16)X[(size_t)b * TT * NIN + (size_t)t * NIN + in_i];
}

__global__ void esn_zero_f16(_Float16* __restrict__ p, int n) {
    int i = blockIdx.x * blockDim.x + threadIdx.x;
    if (i < n) p[i] = (_Float16)0.0f;
}

// ---------------- fragment load at immediate offset ----------------
// 16-bit WMMA A/B per-lane layout (ISA 7.12.2): lane L holds row (L & 15);
// element j of v16h is k = kc + (j/8)*16 + (L/16)*8 + (j%8).
// With p pre-offset by (L/16)*8 elements, that is two 16-byte runs at
// p + kc and p + kc + 16  ->  two global_load_b128 with immediate offsets.
__device__ __forceinline__ v16h esn_frag_at(const _Float16* __restrict__ p, int kc) {
    v8h lo = *(const v8h*)(p + kc);
    v8h hi = *(const v8h*)(p + kc + 16);
    return __builtin_shufflevector(lo, hi, 0, 1, 2, 3, 4, 5, 6, 7,
                                           8, 9, 10, 11, 12, 13, 14, 15);
}

// ---------------- one recurrence step ----------------
// s_next[m, n] = tanh( sum_k Xt[m,k]*Winb[n,k] + sum_k sprev[m,k]*Wb[n,k] )
__global__ void __launch_bounds__(256)
esn_step(const _Float16* __restrict__ Xt,     // [B, 64]
         const _Float16* __restrict__ sprev,  // [B, R]
         _Float16* __restrict__ snext,        // [B, R]
         const _Float16* __restrict__ Winb,   // [R, 64]
         const _Float16* __restrict__ Wb,     // [R, R]
         float* __restrict__ out,             // [B, R] (written when is_last)
         int is_last) {
    const int lane = threadIdx.x & 31;
    const int wave = threadIdx.x >> 5;
    const int wm   = wave & 3;                // 4 M-waves
    const int wn   = wave >> 2;               // 2 N-waves
    const int m0   = blockIdx.y * 64 + wm * 16;
    const int n0   = blockIdx.x * 128 + wn * 64;
    const int lrow = lane & 15;
    const int kh8  = ((lane >> 4) & 1) * 8;   // per-lane K-half pre-offset

    v8f acc[4];
#pragma unroll
    for (int i = 0; i < 4; ++i) acc[i] = (v8f){};

    // ---- input projection: K = 64 over Winb, A rows from Xt ----
    {
        const _Float16* pA  = Xt   + (size_t)(m0 + lrow) * NIN + kh8;
        const _Float16* pB0 = Winb + (size_t)(n0 +  0 + lrow) * NIN + kh8;
        const _Float16* pB1 = Winb + (size_t)(n0 + 16 + lrow) * NIN + kh8;
        const _Float16* pB2 = Winb + (size_t)(n0 + 32 + lrow) * NIN + kh8;
        const _Float16* pB3 = Winb + (size_t)(n0 + 48 + lrow) * NIN + kh8;
#pragma unroll
        for (int kc = 0; kc < NIN; kc += 32) {
            v16h a = esn_frag_at(pA, kc);
            acc[0] = __builtin_amdgcn_wmma_f32_16x16x32_f16(false, a, false, esn_frag_at(pB0, kc), (short)0, acc[0], false, false);
            acc[1] = __builtin_amdgcn_wmma_f32_16x16x32_f16(false, a, false, esn_frag_at(pB1, kc), (short)0, acc[1], false, false);
            acc[2] = __builtin_amdgcn_wmma_f32_16x16x32_f16(false, a, false, esn_frag_at(pB2, kc), (short)0, acc[2], false, false);
            acc[3] = __builtin_amdgcn_wmma_f32_16x16x32_f16(false, a, false, esn_frag_at(pB3, kc), (short)0, acc[3], false, false);
        }
    }

    // ---- recurrent term: K = 2048 over Wb, A rows from sprev ----
    {
        const _Float16* qA  = sprev + (size_t)(m0 + lrow) * RR + kh8;
        const _Float16* qB0 = Wb + (size_t)(n0 +  0 + lrow) * RR + kh8;
        const _Float16* qB1 = Wb + (size_t)(n0 + 16 + lrow) * RR + kh8;
        const _Float16* qB2 = Wb + (size_t)(n0 + 32 + lrow) * RR + kh8;
        const _Float16* qB3 = Wb + (size_t)(n0 + 48 + lrow) * RR + kh8;
#pragma unroll 4
        for (int kc = 0; kc < RR; kc += 32) {
            v16h a = esn_frag_at(qA, kc);
            acc[0] = __builtin_amdgcn_wmma_f32_16x16x32_f16(false, a, false, esn_frag_at(qB0, kc), (short)0, acc[0], false, false);
            acc[1] = __builtin_amdgcn_wmma_f32_16x16x32_f16(false, a, false, esn_frag_at(qB1, kc), (short)0, acc[1], false, false);
            acc[2] = __builtin_amdgcn_wmma_f32_16x16x32_f16(false, a, false, esn_frag_at(qB2, kc), (short)0, acc[2], false, false);
            acc[3] = __builtin_amdgcn_wmma_f32_16x16x32_f16(false, a, false, esn_frag_at(qB3, kc), (short)0, acc[3], false, false);
        }
    }

    // ---- epilogue: tanh, write f16 state (+ f32 output on last step) ----
    // D layout: lane holds N = n_tile + (lane & 15); VGPR v holds M = (lane/16)*8 + v.
    const int mbase = m0 + ((lane >> 4) & 1) * 8;
#pragma unroll
    for (int i = 0; i < 4; ++i) {
        const int n = n0 + i * 16 + lrow;
#pragma unroll
        for (int v = 0; v < 8; ++v) {
            float tv = tanhf(acc[i][v]);
            size_t idx = (size_t)(mbase + v) * RR + n;
            snext[idx] = (_Float16)tv;
            if (is_last) out[idx] = tv;
        }
    }
}

// ---------------- host-side launcher ----------------

extern "C" void kernel_launch(void* const* d_in, const int* in_sizes, int n_in,
                              void* d_out, int out_size, void* d_ws, size_t ws_size,
                              hipStream_t stream) {
    const float* X   = (const float*)d_in[0];   // [B, T, IN]
    const float* Win = (const float*)d_in[1];   // [R, IN]
    const float* W   = (const float*)d_in[2];   // [R, R]
    float* out       = (float*)d_out;           // [B, R]

    char* ws = (char*)d_ws;
    _Float16* Winb = (_Float16*)(ws);                                           // R*IN
    _Float16* Wb   = (_Float16*)(ws + 262144);                                  // R*R
    _Float16* Xb   = (_Float16*)(ws + 262144 + 8388608);                        // T*B*IN
    _Float16* s0   = (_Float16*)(ws + 262144 + 8388608 + 16777216);             // B*R
    _Float16* s1   = (_Float16*)(ws + 262144 + 8388608 + 16777216 + 1048576);   // B*R

    const int nWin = RR * NIN;        // 131072
    const int nW   = RR * RR;         // 4194304
    const int nX   = TT * BB * NIN;   // 8388608
    const int nS   = BB * RR;         // 524288

    esn_cvt_f16<<<(nWin + 255) / 256, 256, 0, stream>>>(Win, Winb, nWin);
    esn_cvt_f16<<<(nW + 255) / 256, 256, 0, stream>>>(W, Wb, nW);
    esn_transpose_x<<<(nX + 255) / 256, 256, 0, stream>>>(X, Xb, nX);
    esn_zero_f16<<<(nS + 255) / 256, 256, 0, stream>>>(s0, nS);

    dim3 grid(RR / 128, BB / 64);   // (16, 4)
    dim3 block(256);

    _Float16* sp = s0;
    _Float16* sn = s1;
    for (int t = 0; t < TT; ++t) {
        const _Float16* Xt = Xb + (size_t)t * BB * NIN;
        esn_step<<<grid, block, 0, stream>>>(Xt, sp, sn, Winb, Wb, out, (t == TT - 1) ? 1 : 0);
        _Float16* tmp = sp; sp = sn; sn = tmp;
    }
}
